// EarthAttention3D_83202106458385
// MI455X (gfx1250) — matrix operand order repaired and checked by reading, not compiled, not run on hardware
//
#include <hip/hip_runtime.h>
#include <hip/hip_bf16.h>

typedef __attribute__((ext_vector_type(16))) _Float16 v16h;
typedef __attribute__((ext_vector_type(8)))  _Float16 v8h;
typedef __attribute__((ext_vector_type(8)))  float    v8f;
typedef __attribute__((ext_vector_type(4)))  unsigned int u32x4;
typedef __attribute__((ext_vector_type(4)))  int      i32x4;
typedef __attribute__((ext_vector_type(8)))  int      i32x8;

// Problem constants
constexpr int N_TOK  = 144;
constexpr int NPAD   = 160;     // padded K-dim for PV gemm (5 x 32)
constexpr int DIM    = 192;
constexpr int HEADS  = 6;
constexpr int HD     = 32;
constexpr int NW     = 840;
constexpr int BATCH  = 2;
constexpr int NBW    = BATCH * NW;              // 1680 windows
constexpr float SCALE = 0.17677669529663687f;   // 32^-0.5

// Workspace: only f16 weights now
constexpr size_t W1H_HALVES = (size_t)3 * DIM * DIM;   // 110592
constexpr size_t W2H_HALVES = (size_t)DIM * DIM;       //  36864
constexpr size_t OFF_W1H = 0;
constexpr size_t OFF_W2H = OFF_W1H + W1H_HALVES * 2;   // 221184

// Dynamic LDS layout (bytes):
constexpr int XS       = 200;                       // x/O LDS stride (400B, 16B-aligned)
constexpr int LDS_SX   = 0;                         // f16 x tile, later O tile: 144*200*2 = 57600
constexpr int LDS_SQ   = 57600;                     // 6*144*32*2 = 55296
constexpr int LDS_SK   = 112896;                    // 55296
constexpr int LDS_SV   = 168192;                    // 6*160*32*2 = 61440
constexpr int LDS_SP   = 229632;                    // 9*16*160*2 = 46080
constexpr int LDS_TOTAL = 275712;                   // < 320KB WGP LDS
// fp32 TDM staging for x overlaps SQ+SK exactly: 144*192*4 = 110592 B
constexpr int LDS_X32  = LDS_SQ;

// ---------------- fragment helpers (wave32 WMMA layouts) ----------------
// A frag (16x32 f16): lane L -> row M = L%16, g = L/16.
// elem e -> K = g*16 + e (contiguous)
__device__ inline v16h load_a_frag(const _Float16* base, int stride, int row,
                                   int col0, int g) {
  const _Float16* p = base + (size_t)row * stride + col0 + g * 16;
  v8h c0 = *reinterpret_cast<const v8h*>(p);
  v8h c1 = *reinterpret_cast<const v8h*>(p + 8);
  v16h a;
#pragma unroll
  for (int i = 0; i < 8; ++i) { a[i] = c0[i]; a[8 + i] = c1[i]; }
  return a;
}

// B frag (32x16 f16) from token-major [n_token][K]: lane col = L%16, elem e -> K = g*16+e
__device__ inline v16h load_b_tokmajor(const _Float16* base, int stride,
                                       int tok, int col0, int g) {
  const _Float16* p = base + (size_t)tok * stride + col0 + g * 16;
  v8h c0 = *reinterpret_cast<const v8h*>(p);
  v8h c1 = *reinterpret_cast<const v8h*>(p + 8);
  v16h b;
#pragma unroll
  for (int i = 0; i < 8; ++i) { b[i] = c0[i]; b[8 + i] = c1[i]; }
  return b;
}

// B frag from K-major [K][ncols]: elem e -> row = row0 + g*16 + e
__device__ inline v16h load_b_kmajor(const _Float16* base, int stride,
                                     int row0, int col, int g) {
  v16h b;
#pragma unroll
  for (int e = 0; e < 16; ++e)
    b[e] = base[(size_t)(row0 + g * 16 + e) * stride + col];
  return b;
}

// ---------------- kernel 0: weight conversion fp32 -> f16 ----------------
__global__ __launch_bounds__(256) void conv_weights(const float* __restrict__ w1,
                                                    const float* __restrict__ w2,
                                                    _Float16* __restrict__ w1h,
                                                    _Float16* __restrict__ w2h) {
  int i = blockIdx.x * blockDim.x + threadIdx.x;
  if (i < (int)W1H_HALVES) w1h[i] = (_Float16)w1[i];
  int j = i - (int)W1H_HALVES;
  if (j >= 0 && j < (int)W2H_HALVES) w2h[j] = (_Float16)w2[j];
}

// ---------------- fused kernel: one block per (b,w) window -------------
// 288 threads = 9 waves (wave32); gemm tiles strided over waves; one 16-row M-tile per wave in attention.
__global__ __launch_bounds__(288) void earth_attn_fused(
    const float* __restrict__ x, const float* __restrict__ mask,
    const _Float16* __restrict__ w1h, const float* __restrict__ b1,
    const _Float16* __restrict__ w2h, const float* __restrict__ b2,
    const float* __restrict__ bt, const int* __restrict__ pi,
    float* __restrict__ out) {
  extern __shared__ char smem[];
  _Float16* sx  = reinterpret_cast<_Float16*>(smem + LDS_SX);
  _Float16* sq  = reinterpret_cast<_Float16*>(smem + LDS_SQ);
  _Float16* sk  = reinterpret_cast<_Float16*>(smem + LDS_SK);
  _Float16* sv  = reinterpret_cast<_Float16*>(smem + LDS_SV);
  _Float16* sp  = reinterpret_cast<_Float16*>(smem + LDS_SP);
  float*    x32 = reinterpret_cast<float*>(smem + LDS_X32);  // TDM staging

  const int bw   = blockIdx.x;
  const int b    = bw / NW;
  const int w    = bw % NW;
  const int tid  = threadIdx.x;
  const int wave = tid >> 5;
  const int lane = tid & 31;
  const int l16  = lane & 15;
  const int g    = lane >> 4;

  // L2 warm-up hints for the (small, shared) f16 weights
  if (tid < 64) __builtin_prefetch(w1h + (size_t)tid * (W1H_HALVES / 64), 0, 1);

  // ---- stage x window into LDS via Tensor Data Mover (2D D#) ----
  if (wave == 0) {
    const unsigned long long ga =
        (unsigned long long)(__SIZE_TYPE__)(x + (size_t)bw * N_TOK * DIM);
    u32x4 g0 = {1u,                                   // count=1 (valid), user mode
                (unsigned)LDS_X32,                    // lds_addr
                (unsigned)(ga & 0xffffffffull),       // global_addr[31:0]
                (unsigned)((ga >> 32) & 0x1ffffffull) | 0x80000000u};  // addr[56:32] | type=2
    i32x8 g1 = {0x20000,                              // data_size=2 (4 bytes)
                (int)(DIM << 16),                     // tensor_dim0 = 192 (bits 63:48)
                (int)(N_TOK << 16),                   // tensor_dim1 = 144 (bits 95:80)
                (int)(DIM << 16),                     // tile_dim0   = 192 (bits 127:112)
                N_TOK,                                // tile_dim1   = 144
                DIM,                                  // tensor_dim0_stride = 192
                0, 0};
    i32x4 z4 = {0, 0, 0, 0};
#if __clang_major__ >= 23
    i32x8 z8 = {0, 0, 0, 0, 0, 0, 0, 0};
    __builtin_amdgcn_tensor_load_to_lds(g0, g1, z4, z4, z8, 0);
#else
    __builtin_amdgcn_tensor_load_to_lds(g0, g1, z4, z4, 0);
#endif
    __builtin_amdgcn_s_wait_tensorcnt(0);
  }
  __syncthreads();

  // ---- convert x fp32 -> f16 (padded stride), zero V pad rows ----
  for (int i = tid; i < N_TOK * DIM; i += 288) {
    int n = i / DIM, c = i % DIM;
    sx[n * XS + c] = (_Float16)x32[i];
  }
  for (int i = tid; i < HEADS * 16 * HD; i += 288) {   // V rows 144..159 = 0
    int h = i / (16 * HD), j = i % (16 * HD);
    sv[(h * NPAD + N_TOK + j / HD) * HD + (j % HD)] = (_Float16)0.0f;
  }
  __syncthreads();

  // ---- phase 1: QKV gemm (overwrites x32 staging with Q/K) ----
  for (int t = wave; t < 9 * 36; t += 9) {
    const int mt = t / 36, ot = t % 36;
    v8f acc = {};
#pragma unroll
    for (int kt = 0; kt < 6; ++kt) {
      v16h a = load_a_frag(sx, XS, mt * 16 + l16, kt * 32, g);
      v16h bf = load_b_tokmajor(w1h, DIM, ot * 16 + l16, kt * 32, g);
      acc = __builtin_amdgcn_wmma_f32_16x16x32_f16(false, a, false, bf,
                                                   (short)0, acc, false, false);
    }
    const int o = ot * 16 + l16;
    const float bb = b1[o];
    const int region = o / DIM;
    const int h  = (o % DIM) / HD;
    const int dd = o % HD;
    _Float16* base = (region == 0) ? sq : (region == 1) ? sk : sv;
    const int hs = (region == 2) ? NPAD : N_TOK;
#pragma unroll
    for (int r = 0; r < 8; ++r) {
      int n = mt * 16 + r + g * 8;
      base[(h * hs + n) * HD + dd] = (_Float16)(acc[r] + bb);
    }
  }
  __syncthreads();

  // ---- phase 2: attention, one head at a time (no block syncs needed) ----
  float rinv[8];
  for (int h = 0; h < HEADS; ++h) {
    v16h aq = load_a_frag(sq + h * N_TOK * HD, HD, wave * 16 + l16, 0, g);
    v8f acc[9];
#pragma unroll
    for (int nt = 0; nt < 9; ++nt) acc[nt] = (v8f){};
#pragma unroll
    for (int nt = 0; nt < 9; ++nt) {
      v16h bk = load_b_tokmajor(sk + h * N_TOK * HD, HD, nt * 16 + l16, 0, g);
      acc[nt] = __builtin_amdgcn_wmma_f32_16x16x32_f16(false, aq, false, bk,
                                                       (short)0, acc[nt],
                                                       false, false);
    }

    // scale + earth bias gather + mask, row softmax (16-lane shfl reductions)
#pragma unroll
    for (int r = 0; r < 8; ++r) {
      const int nq = wave * 16 + r + g * 8;
      float rmax = -3.0e38f;
#pragma unroll
      for (int nt = 0; nt < 9; ++nt) {
        const int mk = nt * 16 + l16;
        const int p  = pi[nq * N_TOK + mk];
        float s = acc[nt][r] * SCALE +
                  bt[((size_t)p * NW + w) * HEADS + h] +
                  mask[((size_t)b * N_TOK + nq) * N_TOK + mk];
        acc[nt][r] = s;
        rmax = fmaxf(rmax, s);
      }
#pragma unroll
      for (int off = 1; off < 16; off <<= 1)
        rmax = fmaxf(rmax, __shfl_xor(rmax, off, 32));
      float rsum = 0.0f;
#pragma unroll
      for (int nt = 0; nt < 9; ++nt) {
        float e = __expf(acc[nt][r] - rmax);
        acc[nt][r] = e;
        rsum += e;
      }
#pragma unroll
      for (int off = 1; off < 16; off <<= 1)
        rsum += __shfl_xor(rsum, off, 32);
      rinv[r] = 1.0f / rsum;
    }

    // P -> per-wave LDS (f16, K padded to 160)
    _Float16* pw = sp + wave * 16 * NPAD;
#pragma unroll
    for (int r = 0; r < 8; ++r) {
      const int rr = r + g * 8;
#pragma unroll
      for (int nt = 0; nt < 9; ++nt)
        pw[rr * NPAD + nt * 16 + l16] = (_Float16)acc[nt][r];
      pw[rr * NPAD + N_TOK + l16] = (_Float16)0.0f;
    }

    // O = P V
    v8f oacc[2];
    oacc[0] = (v8f){};
    oacc[1] = (v8f){};
#pragma unroll
    for (int kt = 0; kt < 5; ++kt) {
      v16h ap = load_a_frag(pw, NPAD, l16, kt * 32, g);
#pragma unroll
      for (int ont = 0; ont < 2; ++ont) {
        v16h bv = load_b_kmajor(sv + h * NPAD * HD, HD, kt * 32,
                                ont * 16 + l16, g);
        oacc[ont] = __builtin_amdgcn_wmma_f32_16x16x32_f16(false, ap, false, bv,
                                                           (short)0, oacc[ont],
                                                           false, false);
      }
    }

    // normalize, store O into sx (x tile no longer needed)
#pragma unroll
    for (int ont = 0; ont < 2; ++ont) {
#pragma unroll
      for (int r = 0; r < 8; ++r) {
        const int nq = wave * 16 + r + g * 8;
        sx[nq * XS + h * HD + ont * 16 + l16] =
            (_Float16)(oacc[ont][r] * rinv[r]);
      }
    }
  }
  __syncthreads();

  // ---- phase 3: output projection, write fp32 ----
  for (int t = wave; t < 9 * 12; t += 9) {
    const int mt = t / 12, nt = t % 12;
    v8f acc = {};
#pragma unroll
    for (int kt = 0; kt < 6; ++kt) {
      v16h a  = load_a_frag(sx, XS, mt * 16 + l16, kt * 32, g);
      v16h bf = load_b_tokmajor(w2h, DIM, nt * 16 + l16, kt * 32, g);
      acc = __builtin_amdgcn_wmma_f32_16x16x32_f16(false, a, false, bf,
                                                   (short)0, acc, false, false);
    }
    const int o = nt * 16 + l16;
    const float bb = b2[o];
#pragma unroll
    for (int r = 0; r < 8; ++r) {
      int n = mt * 16 + r + g * 8;
      out[((size_t)bw * N_TOK + n) * DIM + o] = acc[r] + bb;
    }
  }
}

// ---------------- host launcher ----------------
extern "C" void kernel_launch(void* const* d_in, const int* in_sizes, int n_in,
                              void* d_out, int out_size, void* d_ws, size_t ws_size,
                              hipStream_t stream) {
  const float* x    = (const float*)d_in[0];
  const float* mask = (const float*)d_in[1];
  const float* w1   = (const float*)d_in[2];
  const float* b1   = (const float*)d_in[3];
  const float* w2   = (const float*)d_in[4];
  const float* b2   = (const float*)d_in[5];
  const float* bt   = (const float*)d_in[6];
  const int*   pi   = (const int*)d_in[7];
  float* out        = (float*)d_out;

  char* ws = (char*)d_ws;
  _Float16* w1h = (_Float16*)(ws + OFF_W1H);
  _Float16* w2h = (_Float16*)(ws + OFF_W2H);

  conv_weights<<<(int)((W1H_HALVES + W2H_HALVES + 255) / 256), 256, 0, stream>>>(
      w1, w2, w1h, w2h);
  earth_attn_fused<<<NBW, 288, LDS_TOTAL, stream>>>(x, mask, w1h, b1, w2h, b2,
                                                    bt, pi, out);
}
